// DiagonalSSMLayer_62302795596569
// MI455X (gfx1250) — matrix operand order repaired and checked by reading, not compiled, not run on hardware
//
#include <hip/hip_runtime.h>

// Diagonal SSM layer for gfx1250 (MI455X).
// FFT conv in the reference == first-order recurrence h[l] = lam*h[l-1] + Bu[l].
// Pipeline: [swizzle weights -> bf16 WMMA frag layout] ->
//           [GEMM1 (WMMA bf16, explicit 2-stage SW pipeline) + local scan] ->
//           [chunk-carry scan] ->
//           [carry fixup + GEMM2 (WMMA bf16, 2-stage SW pipeline) + D*u]

#define DM    1024          // d_model
#define NS    256           // state dim
#define BATCH 4
#define SEQ   4096
#define T     64            // chunk rows
#define NCH   (SEQ / T)     // 64 chunks
#define KT1   (DM / 32)     // 32 K-tiles, GEMM1
#define KT2   (NS / 32)     // 8  K-tiles, GEMM2
#define FRAG  512           // elements per (tile,ktile) fragment block: 32 lanes * 16

typedef __attribute__((ext_vector_type(16))) __bf16 v16bf;
typedef __attribute__((ext_vector_type(8)))  float  v8f;

__device__ __forceinline__ float sigmoidf_(float x) {
  return 1.0f / (1.0f + __expf(-x));
}

// A-fragment from fp32 row-major (CDNA5 16-bit A layout, wave32):
// lane row = lane%16, kbase = (lane<16)?0:8,
// elements 0..7 = K kbase..+7, elements 8..15 = K kbase+16..+23.
__device__ __forceinline__ v16bf ldA_f32(const float* p) {
  v16bf r;
#pragma unroll
  for (int g = 0; g < 2; ++g) {
    const float* q = p + g * 16;
#pragma unroll
    for (int i = 0; i < 8; ++i) r[g * 8 + i] = (__bf16)q[i];
  }
  return r;
}

__device__ __forceinline__ v16bf ldA_lds(const __bf16* p) {
  v16bf r;
#pragma unroll
  for (int g = 0; g < 2; ++g) {
#pragma unroll
    for (int i = 0; i < 8; ++i) r[g * 8 + i] = p[g * 16 + i];
  }
  return r;
}

// ==========================================================================
// Kernel 0a/0b: weights fp32 -> bf16, pre-swizzled into the exact B-fragment
// layout frag[tileN][tileK][lane][e]: lane holds n = tileN*16 + (lane&15),
// k = tileK*32 + ((lane&16)?8:0) + (e&7) + ((e&8)?16:0). Runtime frag load = one 32B v16bf.
// ==========================================================================
__global__ void __launch_bounds__(256)
swizzle_Bw(const float* __restrict__ Bw, __bf16* __restrict__ out) {
  int idx = blockIdx.x * 256 + threadIdx.x;          // < 16*32*512 = 262144
  int e    = idx & 15;
  int lane = (idx >> 4) & 31;
  int kt   = (idx >> 9) & (KT1 - 1);
  int t    = idx >> 14;
  int n = t * 16 + (lane & 15);
  int k = kt * 32 + ((lane & 16) ? 8 : 0) + (e & 7) + ((e & 8) ? 16 : 0);
  out[idx] = (__bf16)Bw[(size_t)n * DM + k];
}

__global__ void __launch_bounds__(256)
swizzle_Cw(const float* __restrict__ Cw, __bf16* __restrict__ out) {
  int idx = blockIdx.x * 256 + threadIdx.x;          // < 64*8*512 = 262144
  int e    = idx & 15;
  int lane = (idx >> 4) & 31;
  int kt   = (idx >> 9) & (KT2 - 1);
  int t    = idx >> 12;
  int d = t * 16 + (lane & 15);
  int n = kt * 32 + ((lane & 16) ? 8 : 0) + (e & 7) + ((e & 8) ? 16 : 0);
  out[idx] = (__bf16)Cw[(size_t)d * NS + n];
}

// ==========================================================================
// Kernel 1: Bu = u @ B_w^T (bf16 WMMA) fused with per-chunk local scan.
// One WG (8 waves) per (b, chunk). Explicit 2-stage software pipeline:
// loads of tile k+1 are fenced (sched_barrier) above the WMMAs of tile k.
// ==========================================================================
__global__ void __launch_bounds__(256)
ssm_gemm1_scan(const float* __restrict__ u, const __bf16* __restrict__ BwSw,
               const float* __restrict__ log_lambda,
               float* __restrict__ hloc, float* __restrict__ E) {
  __shared__ float bu[T][NS];  // 64 KB
  const int wg = blockIdx.x;
  const int b  = wg / NCH;
  const int ch = wg % NCH;
  const int tid  = threadIdx.x;
  const int wave = tid >> 5;
  const int lane = tid & 31;
  const int rowTile = wave >> 1;        // 0..3  (16 rows each)
  const int colBase = (wave & 1) * 8;   // 8 col-tiles of 16 -> 128 cols

  const float* aptr = u + ((size_t)b * SEQ + (size_t)ch * T + rowTile * 16) * DM
                        + (size_t)(lane & 15) * DM + ((lane & 16) ? 8 : 0);
  const __bf16* bptr = BwSw + (size_t)colBase * KT1 * FRAG + (size_t)lane * 16;

  v8f acc[8];
#pragma unroll
  for (int t = 0; t < 8; ++t)
#pragma unroll
    for (int j = 0; j < 8; ++j) acc[t][j] = 0.0f;

  v16bf a0, a1;
  v16bf b0[8], b1[8];

  auto loadB = [&](v16bf (&buf)[8], int kt) {
#pragma unroll
    for (int t = 0; t < 8; ++t)
      buf[t] = *(const v16bf*)(bptr + ((size_t)t * KT1 + kt) * FRAG);
  };
  auto fire = [&](const v16bf& a, v16bf (&bb)[8]) {
#pragma unroll
    for (int t = 0; t < 8; ++t)
      acc[t] = __builtin_amdgcn_wmma_f32_16x16x32_bf16(
          false, a, false, bb[t], (short)0, acc[t], false, false);
  };

  a0 = ldA_f32(aptr);
  loadB(b0, 0);
  for (int kt = 0; kt < KT1; kt += 2) {
    // stage 1: prefetch tile kt+1 into buffer 1, compute tile kt on buffer 0
    __builtin_prefetch(aptr + (size_t)((kt + 4 < KT1) ? kt + 4 : kt) * 32, 0, 3);
    a1 = ldA_f32(aptr + (size_t)(kt + 1) * 32);
    loadB(b1, kt + 1);
    __builtin_amdgcn_sched_barrier(0);
    fire(a0, b0);
    // stage 2: prefetch tile kt+2 into buffer 0, compute tile kt+1 on buffer 1
    const int k2 = (kt + 2 < KT1) ? kt + 2 : kt + 1;  // clamp (redundant reload)
    a0 = ldA_f32(aptr + (size_t)k2 * 32);
    loadB(b0, k2);
    __builtin_amdgcn_sched_barrier(0);
    fire(a1, b1);
  }

  // C/D layout: lane col n = lane%16, VGPR i -> row i + (lane<16?0:8)
  const int m0 = (lane & 16) ? 8 : 0;
  const int nn = lane & 15;
#pragma unroll
  for (int t = 0; t < 8; ++t) {
    const int col = (colBase + t) * 16 + nn;
#pragma unroll
    for (int i = 0; i < 8; ++i) bu[rowTile * 16 + m0 + i][col] = acc[t][i];
  }
  __syncthreads();

  // local scan: thread tid owns column n = tid (coalesced/conflict-free)
  const float lam = sigmoidf_(log_lambda[tid]);
  float hprev = 0.0f;
#pragma unroll 4
  for (int l = 0; l < T; ++l) {
    hprev = fmaf(lam, hprev, bu[l][tid]);
    bu[l][tid] = hprev;
  }
  E[((size_t)b * NCH + ch) * NS + tid] = hprev;
  __syncthreads();

  float* out = hloc + ((size_t)b * SEQ + (size_t)ch * T) * NS;
  for (int idx = tid; idx < T * NS; idx += 256) {
    out[idx] = bu[idx >> 8][idx & 255];
  }
}

// ==========================================================================
// Kernel 2: 64-step scan over chunk carries: c[ch] = lam^T*c[ch-1] + E[ch-1]
// ==========================================================================
__global__ void __launch_bounds__(256)
ssm_chunk_scan(const float* __restrict__ E, const float* __restrict__ log_lambda,
               float* __restrict__ carry) {
  const int b = blockIdx.x;
  const int n = threadIdx.x;
  const float lam = sigmoidf_(log_lambda[n]);
  float lamT = lam;
#pragma unroll
  for (int i = 0; i < 6; ++i) lamT *= lamT;  // lam^64 (T = 64)
  float c = 0.0f;
  for (int ch = 0; ch < NCH; ++ch) {
    const size_t o = ((size_t)b * NCH + ch) * NS + n;
    carry[o] = c;
    c = fmaf(lamT, c, E[o]);
  }
}

// ==========================================================================
// Kernel 3: h = h_local + lam^(li+1)*carry (bf16 into LDS), then
//           y = h @ C_w^T + D*u via WMMA (2-stage SW pipeline).
// ==========================================================================
__global__ void __launch_bounds__(256)
ssm_fix_gemm2(const float* __restrict__ hloc, const float* __restrict__ carry,
              const float* __restrict__ log_lambda,
              const float* __restrict__ u, const __bf16* __restrict__ CwSw,
              const float* __restrict__ Dvec, float* __restrict__ y) {
  constexpr int HSTR = NS + 8;     // bf16 row stride (528 B, 16B aligned)
  __shared__ __bf16 hl[T * HSTR];  // ~33 KB
  const int wg = blockIdx.x;
  const int b  = wg / NCH;
  const int ch = wg % NCH;
  const int tid  = threadIdx.x;

  {  // carry fixup, column n = tid (coalesced loads)
    const int n = tid;
    const float lam = sigmoidf_(log_lambda[n]);
    const float c   = carry[((size_t)b * NCH + ch) * NS + n];
    const float* in = hloc + ((size_t)b * SEQ + (size_t)ch * T) * NS + n;
    float f = lam;  // lam^(li+1)
    for (int l = 0; l < T; ++l) {
      hl[l * HSTR + n] = (__bf16)(in[(size_t)l * NS] + f * c);
      f *= lam;
    }
  }
  __syncthreads();

  const int wave = tid >> 5;
  const int lane = tid & 31;
  const int rowTile = wave >> 1;    // 0..3
  const int halfD   = wave & 1;     // which half of D (32 col-tiles each)
  const int m0 = (lane & 16) ? 8 : 0;
  const int nn = lane & 15;
  const __bf16* aptrL = hl + rowTile * 16 * HSTR
                           + (lane & 15) * HSTR + ((lane & 16) ? 8 : 0);

  for (int grp = 0; grp < 4; ++grp) {
    const int colTile0 = halfD * 32 + grp * 8;
    const __bf16* bptr = CwSw + (size_t)colTile0 * KT2 * FRAG + (size_t)lane * 16;

    v8f acc[8];
#pragma unroll
    for (int t = 0; t < 8; ++t)
#pragma unroll
      for (int j = 0; j < 8; ++j) acc[t][j] = 0.0f;

    v16bf a0, a1;
    v16bf b0[8], b1[8];

    auto loadB = [&](v16bf (&buf)[8], int kt) {
#pragma unroll
      for (int t = 0; t < 8; ++t)
        buf[t] = *(const v16bf*)(bptr + ((size_t)t * KT2 + kt) * FRAG);
    };
    auto fire = [&](const v16bf& a, v16bf (&bb)[8]) {
#pragma unroll
      for (int t = 0; t < 8; ++t)
        acc[t] = __builtin_amdgcn_wmma_f32_16x16x32_bf16(
            false, a, false, bb[t], (short)0, acc[t], false, false);
    };

    a0 = ldA_lds(aptrL);
    loadB(b0, 0);
#pragma unroll
    for (int kt = 0; kt < KT2; kt += 2) {  // 4 double-iterations
      a1 = ldA_lds(aptrL + (kt + 1) * 32);
      loadB(b1, kt + 1);
      __builtin_amdgcn_sched_barrier(0);
      fire(a0, b0);
      const int k2 = (kt + 2 < KT2) ? kt + 2 : kt + 1;
      a0 = ldA_lds(aptrL + k2 * 32);
      loadB(b0, k2);
      __builtin_amdgcn_sched_barrier(0);
      fire(a1, b1);
    }

    // epilogue: y = acc + D*u
#pragma unroll
    for (int t = 0; t < 8; ++t) {
      const int d = (colTile0 + t) * 16 + nn;
      const float Dd = Dvec[d];
      const size_t rowbase = (size_t)b * SEQ + (size_t)ch * T + rowTile * 16 + m0;
#pragma unroll
      for (int i = 0; i < 8; ++i) {
        const size_t off = (rowbase + i) * DM + d;
        y[off] = acc[t][i] + Dd * u[off];
      }
    }
  }
}

extern "C" void kernel_launch(void* const* d_in, const int* in_sizes, int n_in,
                              void* d_out, int out_size, void* d_ws, size_t ws_size,
                              hipStream_t stream) {
  const float* u  = (const float*)d_in[0];
  const float* ll = (const float*)d_in[1];
  const float* Bw = (const float*)d_in[2];
  const float* Cw = (const float*)d_in[3];
  const float* Dv = (const float*)d_in[4];
  float* y = (float*)d_out;

  float* hloc  = (float*)d_ws;                                  // B*SEQ*NS f32 = 16 MB
  float* E     = hloc + (size_t)BATCH * SEQ * NS;               // B*NCH*NS f32
  float* carry = E + (size_t)BATCH * NCH * NS;                  // B*NCH*NS f32
  __bf16* BwSw = (__bf16*)(carry + (size_t)BATCH * NCH * NS);   // 256K bf16 (512 KB)
  __bf16* CwSw = BwSw + (size_t)NS * DM;                        // 256K bf16 (512 KB)

  swizzle_Bw<<<(NS * DM) / 256, 256, 0, stream>>>(Bw, BwSw);
  swizzle_Cw<<<(NS * DM) / 256, 256, 0, stream>>>(Cw, CwSw);
  ssm_gemm1_scan<<<BATCH * NCH, 256, 0, stream>>>(u, BwSw, ll, hloc, E);
  ssm_chunk_scan<<<BATCH, 256, 0, stream>>>(E, ll, carry);
  ssm_fix_gemm2<<<BATCH * NCH, 256, 0, stream>>>(hloc, carry, ll, u, CwSw, Dv, y);
}
